// vit22_tformer_34703335752159
// MI455X (gfx1250) — compile-verified, matrix operand order checked by reading, not run on hardware
//
#include <hip/hip_runtime.h>
#include <hip/hip_bf16.h>
#include <math.h>

// ---------------------------------------------------------------------------
// Problem constants (B,S,H,D) = (4,1024,16,64)
// ---------------------------------------------------------------------------
#define B_    4
#define S_    1024
#define H_    16
#define D_    64
#define DIM_  1024
#define FF_   4096
#define FF2_  8192
#define NTOK  4096   // B*S

typedef __attribute__((ext_vector_type(16))) __bf16 v16bf;
typedef __attribute__((ext_vector_type(8)))  float  v8f;
typedef __attribute__((vector_size(16)))     int    v4i_vs;  // 128-bit payload

// gfx1250 async global->LDS copies (ASYNCcnt-tracked), if the toolchain has them.
#if defined(__has_builtin)
#if __has_builtin(__builtin_amdgcn_global_load_async_to_lds_b128) && \
    __has_builtin(__builtin_amdgcn_s_wait_asynccnt)
#define HAVE_ASYNC_LDS 1
#endif
#endif
#ifndef HAVE_ASYNC_LDS
#define HAVE_ASYNC_LDS 0
#endif

#if HAVE_ASYNC_LDS
// Builtin signature (from clang diagnostics): parameters are pointers to
// 128-bit int vectors in the generic address space, plus imm offset / cpol.
static __device__ __forceinline__ void async_b128(const __bf16* g, __bf16* l) {
  __builtin_amdgcn_global_load_async_to_lds_b128(
      (v4i_vs*)(uintptr_t)g, (v4i_vs*)(uintptr_t)l, 0, 0);
}
#endif

static __device__ __forceinline__ v8f wmma_bf16(v16bf a, v16bf b, v8f c) {
  return __builtin_amdgcn_wmma_f32_16x16x32_bf16(false, a, false, b, (short)0, c,
                                                 false, false);
}

// ---------------------------------------------------------------------------
// RMSNorm: h[NTOK,DIM] f32 -> inner bf16
// ---------------------------------------------------------------------------
__global__ __launch_bounds__(256) void k_rmsnorm(const float* __restrict__ h,
                                                 __bf16* __restrict__ out) {
  int row = blockIdx.x;
  const float* x = h + (size_t)row * DIM_;
  __bf16* o = out + (size_t)row * DIM_;
  float ss = 0.f;
  for (int i = threadIdx.x; i < DIM_; i += 256) { float v = x[i]; ss += v * v; }
  __shared__ float red[8];
  for (int off = 16; off > 0; off >>= 1) ss += __shfl_xor(ss, off, 32);
  int wave = threadIdx.x >> 5, lane = threadIdx.x & 31;
  if (lane == 0) red[wave] = ss;
  __syncthreads();
  if (wave == 0) {
    float t = (lane < 8) ? red[lane] : 0.f;
    for (int off = 4; off > 0; off >>= 1) t += __shfl_xor(t, off, 32);
    if (lane == 0) red[0] = t;
  }
  __syncthreads();
  float r = rsqrtf(red[0] / (float)DIM_ + 1.1920928955078125e-07f);
  for (int i = threadIdx.x; i < DIM_; i += 256) o[i] = (__bf16)(x[i] * r);
}

// ---------------------------------------------------------------------------
// Weight convert+transpose: W[K,N] f32 -> Wt[N,K] bf16  (B-operand layout)
// ---------------------------------------------------------------------------
__global__ __launch_bounds__(256) void k_wt(const float* __restrict__ W,
                                            __bf16* __restrict__ Wt,
                                            int K, int N) {
  size_t idx = (size_t)blockIdx.x * 256u + threadIdx.x;
  if (idx >= (size_t)K * N) return;
  int n = (int)(idx / K), k = (int)(idx % K);
  Wt[idx] = (__bf16)W[(size_t)k * N + n];
}

// ---------------------------------------------------------------------------
// WMMA GEMM: C[M,N] = A[M,K](bf16,row) * Bt[N,K](bf16,row) (+bias)
// Block: 128x128, 8 waves in a 2(M)x4(N) grid, wave tile 64x32 (4x2 frags).
// Double-buffered LDS; K-slab k+1 is copied via GLOBAL_LOAD_ASYNC_TO_LDS
// (ASYNCcnt) while WMMAs consume slab k, when the builtin is available.
// ---------------------------------------------------------------------------
__global__ __launch_bounds__(256) void k_gemm(const __bf16* __restrict__ A,
                                              const __bf16* __restrict__ Bt,
                                              const float* __restrict__ bias,
                                              void* __restrict__ Cout,
                                              int M, int N, int K, int outBf16) {
  int tid = threadIdx.x;
  int wave = tid >> 5, lane = tid & 31;
  int l16 = lane & 15, half = lane >> 4;
  int m0 = blockIdx.y * 128, n0 = blockIdx.x * 128;
  int wm = (wave >> 2) * 64, wn = (wave & 3) * 32;
  v8f zz = {0, 0, 0, 0, 0, 0, 0, 0};
  v8f acc[4][2];
#pragma unroll
  for (int i = 0; i < 4; ++i)
#pragma unroll
    for (int j = 0; j < 2; ++j) acc[i][j] = zz;
  int srow = tid >> 1;          // staged row 0..127
  int skk = (tid & 1) * 16;     // 32B chunk within the 32-wide K slab

#if HAVE_ASYNC_LDS
  __shared__ __bf16 lA[2][128 * 32];   // 2 x 8 KB
  __shared__ __bf16 lB[2][128 * 32];   // 2 x 8 KB
  // Prologue: async-stage slab 0 into buffer 0.
  {
    const __bf16* ga = &A[(size_t)(m0 + srow) * K + skk];
    const __bf16* gb = &Bt[(size_t)(n0 + srow) * K + skk];
    async_b128(ga,     &lA[0][srow * 32 + skk]);
    async_b128(ga + 8, &lA[0][srow * 32 + skk + 8]);
    async_b128(gb,     &lB[0][srow * 32 + skk]);
    async_b128(gb + 8, &lB[0][srow * 32 + skk + 8]);
  }
  __builtin_amdgcn_s_wait_asynccnt(0);
  __syncthreads();
  int buf = 0;
  for (int k0 = 0; k0 < K; k0 += 32) {
    // Issue async copy of the NEXT slab into the other buffer.
    if (k0 + 32 < K) {
      const __bf16* ga = &A[(size_t)(m0 + srow) * K + k0 + 32 + skk];
      const __bf16* gb = &Bt[(size_t)(n0 + srow) * K + k0 + 32 + skk];
      async_b128(ga,     &lA[buf ^ 1][srow * 32 + skk]);
      async_b128(ga + 8, &lA[buf ^ 1][srow * 32 + skk + 8]);
      async_b128(gb,     &lB[buf ^ 1][srow * 32 + skk]);
      async_b128(gb + 8, &lB[buf ^ 1][srow * 32 + skk + 8]);
    }
    // Compute on the current buffer while the async copy is in flight.
    v16bf af[4];
#pragma unroll
    for (int mf = 0; mf < 4; ++mf)
      af[mf] = *(const v16bf*)&lA[buf][(wm + mf * 16 + l16) * 32 + half * 16];
#pragma unroll
    for (int nf = 0; nf < 2; ++nf) {
      v16bf bfrag = *(const v16bf*)&lB[buf][(wn + nf * 16 + l16) * 32 + half * 16];
#pragma unroll
      for (int mf = 0; mf < 4; ++mf)
        acc[mf][nf] = wmma_bf16(af[mf], bfrag, acc[mf][nf]);
    }
    __builtin_amdgcn_s_wait_asynccnt(0);
    __syncthreads();
    buf ^= 1;
  }
#else
  __shared__ __bf16 lA[128 * 32];   // 8 KB
  __shared__ __bf16 lB[128 * 32];   // 8 KB
  for (int k0 = 0; k0 < K; k0 += 32) {
    *(v16bf*)&lA[srow * 32 + skk] =
        *(const v16bf*)&A[(size_t)(m0 + srow) * K + k0 + skk];
    *(v16bf*)&lB[srow * 32 + skk] =
        *(const v16bf*)&Bt[(size_t)(n0 + srow) * K + k0 + skk];
    if (k0 + 32 < K) {  // emits global_prefetch_b8 for next slab
      __builtin_prefetch(&A[(size_t)(m0 + srow) * K + k0 + 32 + skk], 0, 1);
      __builtin_prefetch(&Bt[(size_t)(n0 + srow) * K + k0 + 32 + skk], 0, 1);
    }
    __syncthreads();
    v16bf af[4];
#pragma unroll
    for (int mf = 0; mf < 4; ++mf)
      af[mf] = *(const v16bf*)&lA[(wm + mf * 16 + l16) * 32 + half * 16];
#pragma unroll
    for (int nf = 0; nf < 2; ++nf) {
      v16bf bfrag = *(const v16bf*)&lB[(wn + nf * 16 + l16) * 32 + half * 16];
#pragma unroll
      for (int mf = 0; mf < 4; ++mf)
        acc[mf][nf] = wmma_bf16(af[mf], bfrag, acc[mf][nf]);
    }
    __syncthreads();
  }
#endif

  // Epilogue: C layout -> lane l<16: col=l, rows r; lanes 16-31: rows r+8.
#pragma unroll
  for (int mf = 0; mf < 4; ++mf)
#pragma unroll
    for (int nf = 0; nf < 2; ++nf)
#pragma unroll
      for (int r = 0; r < 8; ++r) {
        int rr = m0 + wm + mf * 16 + (half ? r + 8 : r);
        int cc = n0 + wn + nf * 16 + l16;
        float val = acc[mf][nf][r] + (bias ? bias[cc] : 0.f);
        if (outBf16)
          ((__bf16*)Cout)[(size_t)rr * N + cc] = (__bf16)val;
        else
          ((float*)Cout)[(size_t)rr * N + cc] = val;
      }
}

// ---------------------------------------------------------------------------
// L2-norm + RoPE, in place on bf16 [B,S,H,D]. One 32-thread block per (b,s,h).
// ---------------------------------------------------------------------------
__global__ __launch_bounds__(32) void k_rope(__bf16* __restrict__ x) {
  int bsh = blockIdx.x;                 // (b*S+s)*H + h
  int s = (bsh / H_) % S_;
  __bf16* p = x + (size_t)bsh * D_;
  int d = threadIdx.x;                  // 0..31
  float x1 = (float)p[d], x2 = (float)p[d + 32];
  float ss = x1 * x1 + x2 * x2;
  for (int off = 16; off > 0; off >>= 1) ss += __shfl_xor(ss, off, 32);
  float r = 1.0f / fmaxf(sqrtf(ss), 1e-12f);
  x1 *= r; x2 *= r;
  float invf = __powf(1000.0f, -((float)d / 32.0f));
  float ang = (float)s * invf;
  float cf = (float)(__bf16)__cosf(ang);   // match bf16 rope tables
  float sf = (float)(__bf16)__sinf(ang);
  p[d]      = (__bf16)( x1 * cf + x2 * sf);
  p[d + 32] = (__bf16)(-x1 * sf + x2 * cf);
}

// ---------------------------------------------------------------------------
// brow[b,i,h] = scale * bq_i . prefix_j<=i(bk_j)   (masked GEMM collapsed)
// ---------------------------------------------------------------------------
__global__ void k_prefix(const __bf16* __restrict__ bk, float* __restrict__ pre) {
  int t = blockIdx.x * 256 + threadIdx.x;  // over B*H*D
  if (t >= B_ * H_ * D_) return;
  int d = t % D_, h = (t / D_) % H_, b = t / (D_ * H_);
  float c = 0.f;
  for (int s = 0; s < S_; ++s) {
    size_t idx = ((size_t)(b * S_ + s) * H_ + h) * D_ + d;
    c += (float)bk[idx];
    pre[idx] = c;
  }
}

__global__ __launch_bounds__(32) void k_brow(const __bf16* __restrict__ bq,
                                             const float* __restrict__ pre,
                                             float* __restrict__ brow) {
  int bsh = blockIdx.x;
  int d = threadIdx.x;
  size_t base = (size_t)bsh * D_;
  float acc = (float)bq[base + d] * pre[base + d] +
              (float)bq[base + d + 32] * pre[base + d + 32];
  for (int off = 16; off > 0; off >>= 1) acc += __shfl_xor(acc, off, 32);
  if (d == 0) brow[bsh] = 0.125f * acc;   // D^-0.5 = 1/8
}

// ---------------------------------------------------------------------------
// Flash attention with online softmax. Block = 256 thr (8 waves); each block
// handles (b,h) x 128 query rows; each wave owns 16 rows. QK^T and PV via
// WMMA bf16. Fuses l2normscale and the brow broadcast. Output y bf16 [B,S,DIM].
// ---------------------------------------------------------------------------
__global__ __launch_bounds__(256) void k_attn(const __bf16* __restrict__ q,
                                              const __bf16* __restrict__ k,
                                              const __bf16* __restrict__ v,
                                              const float* __restrict__ brow,
                                              const float* __restrict__ l2s_p,
                                              __bf16* __restrict__ y) {
  __shared__ __bf16 vt[D_ * 32];      // V tile transposed [d][j]  (4 KB)
  __shared__ __bf16 pl[8 * 16 * 32];  // per-wave P staging       (8 KB)
  int tid = threadIdx.x;
  int wave = tid >> 5, lane = tid & 31;
  int l16 = lane & 15, half = lane >> 4;
  int bh = blockIdx.y, b = bh / H_, h = bh % H_;
  int qTile = blockIdx.x * 128;
  int qBase = qTile + wave * 16;
  float l2s = *l2s_p;

  // Q fragments (row = qBase+l16, two K-chunks of 32 head dims).
  v16bf aq[2];
  {
    const __bf16* qp = q + ((size_t)(b * S_ + qBase + l16) * H_ + h) * D_;
    aq[0] = *(const v16bf*)(qp + half * 16);
    aq[1] = *(const v16bf*)(qp + 32 + half * 16);
  }
  v8f zz = {0, 0, 0, 0, 0, 0, 0, 0};
  v8f o[4];
#pragma unroll
  for (int f = 0; f < 4; ++f) o[f] = zz;
  float mstat[8], lstat[8];
#pragma unroll
  for (int r = 0; r < 8; ++r) { mstat[r] = -1e30f; lstat[r] = 0.f; }

  int jEnd = qTile + 128;                        // block-uniform trip count
  for (int jb = 0; jb < jEnd; jb += 32) {
    // Cooperative transpose-stage of V tile: vt[d][j].
    {
      int j = tid >> 3;
      int dc = (tid & 7) * 8;
      const __bf16* vp = v + ((size_t)(b * S_ + jb + j) * H_ + h) * D_ + dc;
#pragma unroll
      for (int i = 0; i < 8; ++i) vt[(dc + i) * 32 + j] = vp[i];
    }
    __syncthreads();
    // Scores: S16x32 = Q(16x64) . K^T via 4 WMMA (2 n-frags x 2 k-chunks).
    v8f s0 = zz, s1 = zz;
#pragma unroll
    for (int c = 0; c < 2; ++c) {
      v16bf bk0 = *(const v16bf*)(k + ((size_t)(b * S_ + jb + l16) * H_ + h) * D_ +
                                  c * 32 + half * 16);
      v16bf bk1 = *(const v16bf*)(k + ((size_t)(b * S_ + jb + 16 + l16) * H_ + h) * D_ +
                                  c * 32 + half * 16);
      s0 = wmma_bf16(aq[c], bk0, s0);
      s1 = wmma_bf16(aq[c], bk1, s1);
    }
    // Causal mask + online softmax update (per-row stats across 16 lanes).
#pragma unroll
    for (int r = 0; r < 8; ++r) {
      int row = qBase + (half ? r + 8 : r);
      int c0 = jb + l16, c1 = c0 + 16;
      float v0 = (c0 <= row) ? s0[r] : -1e30f;
      float v1 = (c1 <= row) ? s1[r] : -1e30f;
      float cm = fmaxf(v0, v1);
      for (int off = 8; off > 0; off >>= 1) cm = fmaxf(cm, __shfl_xor(cm, off, 32));
      float mnew = fmaxf(mstat[r], cm);
      float p0 = __expf(v0 - mnew), p1 = __expf(v1 - mnew);
      float ps = p0 + p1;
      for (int off = 8; off > 0; off >>= 1) ps += __shfl_xor(ps, off, 32);
      float alpha = __expf(mstat[r] - mnew);
      lstat[r] = lstat[r] * alpha + ps;
      mstat[r] = mnew;
#pragma unroll
      for (int f = 0; f < 4; ++f) o[f][r] *= alpha;
      int rl = half ? r + 8 : r;
      pl[wave * 512 + rl * 32 + l16]      = (__bf16)p0;
      pl[wave * 512 + rl * 32 + 16 + l16] = (__bf16)p1;
    }
    __syncthreads();
    // O += P(16x32) . V(32x64) via 4 WMMA.
    v16bf pf = *(const v16bf*)&pl[wave * 512 + l16 * 32 + half * 16];
#pragma unroll
    for (int f = 0; f < 4; ++f) {
      v16bf vf = *(const v16bf*)&vt[(f * 16 + l16) * 32 + half * 16];
      o[f] = wmma_bf16(pf, vf, o[f]);
    }
    __syncthreads();
  }
  // Epilogue: y = l2scale * O / l + brow  (broadcast over head dim).
#pragma unroll
  for (int r = 0; r < 8; ++r) {
    int row = qBase + (half ? r + 8 : r);
    float inv = l2s / lstat[r];
    float br = brow[(size_t)(b * S_ + row) * H_ + h];
#pragma unroll
    for (int f = 0; f < 4; ++f) {
      float val = o[f][r] * inv + br;
      y[(size_t)(b * S_ + row) * DIM_ + h * 64 + f * 16 + l16] = (__bf16)val;
    }
  }
}

// ---------------------------------------------------------------------------
// SwiGLU gate: act = silu(u[:,FF:]) * u[:,:FF]
// ---------------------------------------------------------------------------
__global__ __launch_bounds__(256) void k_silu(const __bf16* __restrict__ u,
                                              __bf16* __restrict__ act) {
  size_t idx = (size_t)blockIdx.x * 256u + threadIdx.x;
  if (idx >= (size_t)NTOK * FF_) return;
  size_t row = idx / FF_, col = idx % FF_;
  float xh = (float)u[row * FF2_ + col];
  float g  = (float)u[row * FF2_ + FF_ + col];
  float sg = g / (1.f + __expf(-g));
  act[idx] = (__bf16)(sg * xh);
}

// ---------------------------------------------------------------------------
// Final residual: out = h*lambda + dense + attn
// ---------------------------------------------------------------------------
__global__ __launch_bounds__(256) void k_final(const float* __restrict__ h,
                                               const float* __restrict__ dense,
                                               const float* __restrict__ attn,
                                               const float* __restrict__ lam_p,
                                               float* __restrict__ out) {
  size_t idx = (size_t)blockIdx.x * 256u + threadIdx.x;
  if (idx >= (size_t)NTOK * DIM_) return;
  out[idx] = h[idx] * (*lam_p) + dense[idx] + attn[idx];
}

// ---------------------------------------------------------------------------
// Host orchestration
// ---------------------------------------------------------------------------
extern "C" void kernel_launch(void* const* d_in, const int* in_sizes, int n_in,
                              void* d_out, int out_size, void* d_ws, size_t ws_size,
                              hipStream_t stream) {
  (void)in_sizes; (void)n_in; (void)out_size; (void)ws_size;
  const float* h_states = (const float*)d_in[0];
  // d_in[1] attention_mask: causal tril, folded analytically.
  const float* Wq      = (const float*)d_in[2];
  const float* Wk      = (const float*)d_in[3];
  const float* Wv      = (const float*)d_in[4];
  const float* WqB     = (const float*)d_in[5];
  const float* WkB     = (const float*)d_in[6];
  const float* Wo      = (const float*)d_in[7];
  const float* bo      = (const float*)d_in[8];
  const float* Wff_in  = (const float*)d_in[9];
  const float* bff_in  = (const float*)d_in[10];
  const float* Wff_out = (const float*)d_in[11];
  const float* bff_out = (const float*)d_in[12];
  const float* lam     = (const float*)d_in[13];
  const float* l2s     = (const float*)d_in[14];
  float* out = (float*)d_out;

  char* ws = (char*)d_ws;
  size_t off = 0;
  auto alloc = [&](size_t bytes) -> void* {
    void* p = ws + off;
    off += (bytes + 255) & ~(size_t)255;
    return p;
  };
  __bf16* innerB   = (__bf16*)alloc((size_t)NTOK * DIM_ * 2);
  __bf16* wq_t     = (__bf16*)alloc((size_t)DIM_ * DIM_ * 2);
  __bf16* wk_t     = (__bf16*)alloc((size_t)DIM_ * DIM_ * 2);
  __bf16* wv_t     = (__bf16*)alloc((size_t)DIM_ * DIM_ * 2);
  __bf16* wqB_t    = (__bf16*)alloc((size_t)DIM_ * DIM_ * 2);
  __bf16* wkB_t    = (__bf16*)alloc((size_t)DIM_ * DIM_ * 2);
  __bf16* wo_t     = (__bf16*)alloc((size_t)DIM_ * DIM_ * 2);
  __bf16* wffin_t  = (__bf16*)alloc((size_t)FF2_ * DIM_ * 2);
  __bf16* wffout_t = (__bf16*)alloc((size_t)DIM_ * FF_ * 2);
  __bf16* qlin     = (__bf16*)alloc((size_t)NTOK * DIM_ * 2);
  __bf16* klin     = (__bf16*)alloc((size_t)NTOK * DIM_ * 2);
  __bf16* vlin     = (__bf16*)alloc((size_t)NTOK * DIM_ * 2);
  __bf16* bqlin    = (__bf16*)alloc((size_t)NTOK * DIM_ * 2);
  __bf16* bklin    = (__bf16*)alloc((size_t)NTOK * DIM_ * 2);
  float*  prefixK  = (float*) alloc((size_t)NTOK * DIM_ * 4);
  float*  browf    = (float*) alloc((size_t)NTOK * H_ * 4);
  __bf16* yb       = (__bf16*)alloc((size_t)NTOK * DIM_ * 2);
  float*  attn_out = (float*) alloc((size_t)NTOK * DIM_ * 4);
  __bf16* ub       = (__bf16*)alloc((size_t)NTOK * FF2_ * 2);
  __bf16* actb     = (__bf16*)alloc((size_t)NTOK * FF_ * 2);
  float*  densef   = (float*) alloc((size_t)NTOK * DIM_ * 4);

  // 1. RMSNorm -> inner (bf16)
  k_rmsnorm<<<NTOK, 256, 0, stream>>>(h_states, innerB);

  // 2. Weight convert + transpose to [N,K] bf16
  {
    int g1 = (int)(((size_t)DIM_ * DIM_ + 255) / 256);
    k_wt<<<g1, 256, 0, stream>>>(Wq,  wq_t,  DIM_, DIM_);
    k_wt<<<g1, 256, 0, stream>>>(Wk,  wk_t,  DIM_, DIM_);
    k_wt<<<g1, 256, 0, stream>>>(Wv,  wv_t,  DIM_, DIM_);
    k_wt<<<g1, 256, 0, stream>>>(WqB, wqB_t, DIM_, DIM_);
    k_wt<<<g1, 256, 0, stream>>>(WkB, wkB_t, DIM_, DIM_);
    k_wt<<<g1, 256, 0, stream>>>(Wo,  wo_t,  DIM_, DIM_);
    int g2 = (int)(((size_t)DIM_ * FF2_ + 255) / 256);
    k_wt<<<g2, 256, 0, stream>>>(Wff_in, wffin_t, DIM_, FF2_);
    int g3 = (int)(((size_t)FF_ * DIM_ + 255) / 256);
    k_wt<<<g3, 256, 0, stream>>>(Wff_out, wffout_t, FF_, DIM_);
  }

  // 3. Projection GEMMs (WMMA): [4096,1024] x [1024,1024] -> bf16
  dim3 gp(DIM_ / 128, NTOK / 128);
  k_gemm<<<gp, 256, 0, stream>>>(innerB, wq_t,  nullptr, qlin,  NTOK, DIM_, DIM_, 1);
  k_gemm<<<gp, 256, 0, stream>>>(innerB, wk_t,  nullptr, klin,  NTOK, DIM_, DIM_, 1);
  k_gemm<<<gp, 256, 0, stream>>>(innerB, wv_t,  nullptr, vlin,  NTOK, DIM_, DIM_, 1);
  k_gemm<<<gp, 256, 0, stream>>>(innerB, wqB_t, nullptr, bqlin, NTOK, DIM_, DIM_, 1);
  k_gemm<<<gp, 256, 0, stream>>>(innerB, wkB_t, nullptr, bklin, NTOK, DIM_, DIM_, 1);

  // 4. L2-norm + RoPE (in place, v untouched)
  int nbsh = B_ * S_ * H_;
  k_rope<<<nbsh, 32, 0, stream>>>(qlin);
  k_rope<<<nbsh, 32, 0, stream>>>(klin);
  k_rope<<<nbsh, 32, 0, stream>>>(bqlin);
  k_rope<<<nbsh, 32, 0, stream>>>(bklin);

  // 5. brow via key prefix sums (masked GEMM collapsed to O(S*H*D))
  k_prefix<<<(B_ * H_ * D_ + 255) / 256, 256, 0, stream>>>(bklin, prefixK);
  k_brow<<<nbsh, 32, 0, stream>>>(bqlin, prefixK, browf);

  // 6. Flash attention (WMMA QK^T + PV, online softmax, brow fused)
  dim3 ga(S_ / 128, B_ * H_);
  k_attn<<<ga, 256, 0, stream>>>(qlin, klin, vlin, browf, l2s, yb);

  // 7. Output projection: y @ Wo + bo -> f32
  k_gemm<<<gp, 256, 0, stream>>>(yb, wo_t, bo, attn_out, NTOK, DIM_, DIM_, 0);

  // 8. FFN in: inner @ Wff_in + bff_in -> bf16 [4096,8192]
  dim3 gf1(FF2_ / 128, NTOK / 128);
  k_gemm<<<gf1, 256, 0, stream>>>(innerB, wffin_t, bff_in, ub, NTOK, FF2_, DIM_, 1);

  // 9. SwiGLU gate
  k_silu<<<(int)(((size_t)NTOK * FF_ + 255) / 256), 256, 0, stream>>>(ub, actb);

  // 10. FFN out: act @ Wff_out + bff_out -> f32
  k_gemm<<<gp, 256, 0, stream>>>(actb, wffout_t, bff_out, densef, NTOK, DIM_, FF_, 0);

  // 11. Residual combine
  k_final<<<(int)(((size_t)NTOK * DIM_ + 255) / 256), 256, 0, stream>>>(
      h_states, densef, attn_out, lam, out);
}
